// TDG4MSF_55147380080738
// MI455X (gfx1250) — matrix operand order repaired and checked by reading, not compiled, not run on hardware
//
#include <hip/hip_runtime.h>
#include <hip/hip_bf16.h>
#include <math.h>

// ---------------------------------------------------------------------------
// Problem constants (match reference)
// ---------------------------------------------------------------------------
#define Bn   32
#define Nn   2048
#define Tn   96
#define Kk   20
#define DN   40
#define DH   64
#define DG   16

static constexpr size_t ROWSC = (size_t)Bn * Nn;          // 65536 flattened (b,n) rows

typedef __attribute__((ext_vector_type(16))) _Float16 v16h;
typedef __attribute__((ext_vector_type(8)))  float    v8f;

// ---------------------------------------------------------------------------
// Workspace layout (bytes). Total ~73 MB.
// ---------------------------------------------------------------------------
static constexpr size_t SZ_ROWT_H = ROWSC * Tn * sizeof(_Float16);   // 12,582,912
static constexpr size_t SZ_ROWT_F = ROWSC * Tn * sizeof(float);      // 25,165,824
static constexpr size_t OFF_SEAS16 = 0;                              // half[ROWS][96]
static constexpr size_t OFF_TRS16  = OFF_SEAS16 + SZ_ROWT_H;         // half[ROWS][96]
static constexpr size_t OFF_XX     = OFF_TRS16  + SZ_ROWT_H;         // float[ROWS][96]
static constexpr size_t OFF_SKIP   = OFF_XX     + SZ_ROWT_F;         // float[ROWS][96]
static constexpr size_t OFF_NV1    = OFF_SKIP   + SZ_ROWT_F;         // float[2048][40]
static constexpr size_t OFF_NV2    = OFF_NV1 + (size_t)Nn*DN*4;
static constexpr size_t OFF_COLS   = OFF_NV2 + (size_t)Nn*DN*4;      // int[2048][20]
static constexpr size_t OFF_MASK   = OFF_COLS + (size_t)Nn*Kk*4;     // float[2048][20]
static constexpr size_t OFF_W      = OFF_MASK + (size_t)Nn*Kk*4;     // f16 weight pool
// f16 weight pool offsets (in halfs), all stored as Bt[k][o] (K-major)
static constexpr size_t W_SKIP0 = 0;
static constexpr size_t W_SEAS  = W_SKIP0 + 96*96;
static constexpr size_t W_TREND = W_SEAS  + 96*96;
static constexpr size_t W_E1    = W_TREND + 96*96;
static constexpr size_t W_LIN1  = W_E1    + 96*96;   // [96][64]
static constexpr size_t W_LIN3  = W_LIN1  + 96*64;   // [96][64]
static constexpr size_t W_LIN2  = W_LIN3  + 96*64;   // [32][96] (rows 16..31 zero)
static constexpr size_t W_MSG   = W_LIN2  + 32*96;   // [64][16]
static constexpr size_t W_GATE  = W_MSG   + 64*16;   // [64][16]
static constexpr size_t W_TOTAL = W_GATE  + 64*16;
static constexpr size_t OFF_STATS = OFF_W + W_TOTAL*sizeof(_Float16); // float[64]
// Aliases into dead regions:
static constexpr size_t OFF_XG16  = OFF_SEAS16;                         // half[ROWS][64]
static constexpr size_t OFF_GOUT  = OFF_SEAS16 + ROWSC*DH*sizeof(_Float16); // half[ROWS][16]
static constexpr size_t OFF_H     = OFF_TRS16;                          // float[ROWS][16]
static constexpr size_t OFF_GATE  = OFF_TRS16 + ROWSC*DG*4;             // float[ROWS][16]
static constexpr size_t OFF_AGG   = OFF_TRS16 + 2*ROWSC*DG*4;           // float[ROWS][16]

// ---------------------------------------------------------------------------
// WMMA fragment-order LDS swizzles (layouts per CDNA5 ISA §7.12.2).
// Tiles are stored in LDS pre-permuted so each lane's 16-half fragment is 32
// contiguous bytes -> two ds_load_b128 per fragment, no v_mov_b16 fixup.
//
// A fragment (16x32 f16): lane(m=lane%16, hi=lane/16) gets K = kb + hi*16 + j (order chosen here, same as B).
// B fragment (32x16 f16): lane(n=lane%16, hi=lane/16) gets K = kb + hi*16 + j (order chosen here, same as A).
// ---------------------------------------------------------------------------
__device__ __forceinline__ int tdg_aswz(int r, int k) {
  int chunk = k >> 5, kk = k & 31;
  int hi = kk >> 4, j = kk & 15;
  return (((chunk << 1) + hi) * 16 + r) * 16 + j;
}

__device__ __forceinline__ int tdg_bswz(int k, int col, int NO_) {
  int chunk = k >> 5, kk = k & 31;
  int hi = kk >> 4, j = kk & 15;
  return (((chunk << 1) + hi) * NO_ + col) * 16 + j;
}

__device__ __forceinline__ v16h tdg_frag_a(const _Float16* sA, int chunk, int lane) {
  int m = lane & 15, hi = lane >> 4;
  return *(const v16h*)(sA + ((((chunk << 1) + hi) * 16 + m) << 4));
}

__device__ __forceinline__ v16h tdg_frag_b(const _Float16* sB, int NO_, int chunk, int cb, int lane) {
  int n = lane & 15, hi = lane >> 4;
  return *(const v16h*)(sB + ((((chunk << 1) + hi) * NO_ + cb + n) << 4));
}

// ---------------------------------------------------------------------------
// Weight prep: dst[bswz(k,o)] = (half)(k < srcK ? W[o*srcK + k] : 0)
// Weights are written directly in B fragment-order for their NO width.
// ---------------------------------------------------------------------------
__global__ void tdg_wprep(const float* __restrict__ W, _Float16* __restrict__ dst,
                          int O, int srcK, int dstK) {
  int i = blockIdx.x * blockDim.x + threadIdx.x;
  if (i >= O * dstK) return;
  int k = i / O, o = i - k * O;
  dst[tdg_bswz(k, o, O)] = (_Float16)(k < srcK ? W[o * srcK + k] : 0.0f);
}

// ---------------------------------------------------------------------------
// nv = tanh(3*(emb @ w.T + b))   [2048][40]
// ---------------------------------------------------------------------------
__global__ void tdg_nv(const float* __restrict__ emb, const float* __restrict__ w,
                       const float* __restrict__ bias, float* __restrict__ out) {
  int i = blockIdx.x * blockDim.x + threadIdx.x;
  if (i >= Nn * DN) return;
  int r = i / DN, j = i - r * DN;
  const float* e = emb + (size_t)r * DN;
  const float* wr = w + (size_t)j * DN;
  float acc = bias[j];
#pragma unroll 8
  for (int d = 0; d < DN; ++d) acc += e[d] * wr[d];
  out[i] = tanhf(3.0f * acc);
}

// ---------------------------------------------------------------------------
// Per-row top-K of relu(tanh(3*(nv1@nv2.T - nv2@nv1.T))). One block per row.
// ---------------------------------------------------------------------------
__global__ void tdg_topk(const float* __restrict__ nv1, const float* __restrict__ nv2,
                         int* __restrict__ cols, float* __restrict__ maskv) {
  int r = blockIdx.x, tid = threadIdx.x;       // blockDim = 256
  __shared__ float s1[DN], s2[DN];
  __shared__ float bval[256];
  __shared__ int   bcol[256];
  if (tid < DN) { s1[tid] = nv1[(size_t)r*DN + tid]; s2[tid] = nv2[(size_t)r*DN + tid]; }
  __syncthreads();
  float vals[8];
#pragma unroll
  for (int i = 0; i < 8; ++i) {
    int c = tid + 256 * i;
    const float* p1 = nv1 + (size_t)c * DN;
    const float* p2 = nv2 + (size_t)c * DN;
    float af = 0.f, ab = 0.f;
#pragma unroll 8
    for (int d = 0; d < DN; ++d) { af += s1[d] * p2[d]; ab += s2[d] * p1[d]; }
    float t = tanhf(3.0f * (af - ab));
    vals[i] = t > 0.f ? t : 0.f;
  }
  for (int sel = 0; sel < Kk; ++sel) {
    float mv = vals[0]; int mc = tid;
#pragma unroll
    for (int i = 1; i < 8; ++i)
      if (vals[i] > mv) { mv = vals[i]; mc = tid + 256 * i; }
    bval[tid] = mv; bcol[tid] = mc;
    __syncthreads();
    for (int s = 128; s > 0; s >>= 1) {
      if (tid < s) {
        float ov = bval[tid + s]; int oc = bcol[tid + s];
        if (ov > bval[tid] || (ov == bval[tid] && oc < bcol[tid])) { bval[tid] = ov; bcol[tid] = oc; }
      }
      __syncthreads();
    }
    int bc = bcol[0]; float bv = bval[0];
    if (tid == 0) { cols[r*Kk + sel] = bc; maskv[r*Kk + sel] = bv > 0.f ? 1.f : 0.f; }
    if ((bc & 255) == tid) vals[bc >> 8] = -1.0f;   // remove selected
    __syncthreads();
  }
}

// ---------------------------------------------------------------------------
// Series decomposition: two window-25 moving averages (replicate pad == clamp).
// One block (96 threads) per (b,n) row. Writes f16 seasonal and trend-sum.
// ---------------------------------------------------------------------------
__global__ void tdg_decomp(const float* __restrict__ x,
                           _Float16* __restrict__ seas16, _Float16* __restrict__ trs16) {
  int row = blockIdx.x, t = threadIdx.x;
  __shared__ float xr[Tn], st[Tn];
  xr[t] = x[(size_t)row * Tn + t];
  __syncthreads();
  float s = 0.f;
#pragma unroll
  for (int u = -12; u <= 12; ++u) {
    int uu = t + u; uu = uu < 0 ? 0 : (uu > Tn-1 ? Tn-1 : uu);
    s += xr[uu];
  }
  float ma1 = s * (1.0f/25.0f);
  st[t] = xr[t] - ma1;
  __syncthreads();
  s = 0.f;
#pragma unroll
  for (int u = -12; u <= 12; ++u) {
    int uu = t + u; uu = uu < 0 ? 0 : (uu > Tn-1 ? Tn-1 : uu);
    s += st[uu];
  }
  float ma2 = s * (1.0f/25.0f);
  seas16[(size_t)row * Tn + t] = (_Float16)(st[t] - ma2);
  trs16[(size_t)row * Tn + t]  = (_Float16)(ma1 + ma2);
}

// ---------------------------------------------------------------------------
// Generic 16-row-block WMMA GEMM: C[16][NO] = A[16][KSRC] (padded to KD) @ Bt[KD][NO]
// Bt is pre-swizzled in fragment order (tdg_bswz with width NO).
// AMODE: 0=f32 A, 1=f16 A, 2=f32 A with relu. OMODE: 0=store f32(+bias),
// 1=accumulate into f32(+bias), 2=store f16(+bias), 3=store f32(+bias+residual).
// blockDim = (NO/16)*32, grid = ROWS/16.
// ---------------------------------------------------------------------------
template<int KD, int KSRC, int NO, int AMODE, int OMODE>
__global__ void tdg_gemm16(const void* __restrict__ Asrc,
                           const _Float16* __restrict__ Bt,
                           const float* __restrict__ bias,
                           float* __restrict__ OutF,
                           _Float16* __restrict__ OutH,
                           const float* __restrict__ Res) {
  constexpr int NW = NO / 16;
  __shared__ alignas(32) _Float16 sA[16 * KD];
  __shared__ alignas(32) _Float16 sB[KD * NO];
  int row0 = blockIdx.x * 16;
  int tid = threadIdx.x, lane = tid & 31, wave = tid >> 5;

  for (int i = tid; i < 16 * KD; i += NW * 32) {
    int r = i / KD, k = i - r * KD;
    float v = 0.f;
    if (k < KSRC) {
      if (AMODE == 1) v = (float)((const _Float16*)Asrc)[(size_t)(row0 + r) * KSRC + k];
      else {
        v = ((const float*)Asrc)[(size_t)(row0 + r) * KSRC + k];
        if (AMODE == 2) v = fmaxf(v, 0.f);
      }
    }
    sA[tdg_aswz(r, k)] = (_Float16)v;
  }
  for (int i = tid; i < KD * NO; i += NW * 32) sB[i] = Bt[i];  // already fragment order
  if (AMODE != 1 && tid < 16)   // warm L2/L0 for the next row-block's A stream
    __builtin_prefetch((const char*)Asrc + (size_t)(row0 + 16 + tid) * KSRC * 4, 0, 1);
  __syncthreads();

  v8f acc = {};
  int cb = wave * 16;
#pragma unroll
  for (int chunk = 0; chunk < KD / 32; ++chunk) {
    v16h a = tdg_frag_a(sA, chunk, lane);
    v16h b = tdg_frag_b(sB, NO, chunk, cb, lane);
    acc = __builtin_amdgcn_wmma_f32_16x16x32_f16(false, a, false, b, (short)0, acc, false, false);
  }

  int n = lane & 15, hi = lane >> 4, col = cb + n;
  float bv = bias[col];
#pragma unroll
  for (int v = 0; v < 8; ++v) {
    size_t idx = (size_t)(row0 + hi*8 + v) * NO + col;
    float val = acc[v] + bv;
    if (OMODE == 0)      OutF[idx] = val;
    else if (OMODE == 1) OutF[idx] += val;
    else if (OMODE == 2) OutH[idx] = (_Float16)val;
    else                 OutF[idx] = val + Res[idx];
  }
}

// ---------------------------------------------------------------------------
// xx = seas@Ws + trs@Wt + (bs+bt): two-pass accumulation, same accumulators.
// ---------------------------------------------------------------------------
__global__ void tdg_xx_gemm(const _Float16* __restrict__ A1, const _Float16* __restrict__ B1,
                            const _Float16* __restrict__ A2, const _Float16* __restrict__ B2,
                            const float* __restrict__ bias1, const float* __restrict__ bias2,
                            float* __restrict__ OutF) {
  constexpr int KD = 96, NO = 96, NW = 6;
  __shared__ alignas(32) _Float16 sA[16 * KD];
  __shared__ alignas(32) _Float16 sB[KD * NO];
  int row0 = blockIdx.x * 16;
  int tid = threadIdx.x, lane = tid & 31, wave = tid >> 5;
  v8f acc = {};
  int cb = wave * 16;
  for (int pass = 0; pass < 2; ++pass) {
    const _Float16* A = pass ? A2 : A1;
    const _Float16* Bm = pass ? B2 : B1;
    if (pass) __syncthreads();                 // prior wmma LDS reads complete
    for (int i = tid; i < 16 * KD; i += NW * 32) {
      int r = i / KD, k = i - r * KD;
      sA[tdg_aswz(r, k)] = A[(size_t)(row0 + r) * KD + k];
    }
    for (int i = tid; i < KD * NO; i += NW * 32) sB[i] = Bm[i];
    __syncthreads();
#pragma unroll
    for (int chunk = 0; chunk < KD / 32; ++chunk) {
      v16h a = tdg_frag_a(sA, chunk, lane);
      v16h b = tdg_frag_b(sB, NO, chunk, cb, lane);
      acc = __builtin_amdgcn_wmma_f32_16x16x32_f16(false, a, false, b, (short)0, acc, false, false);
    }
  }
  int n = lane & 15, hi = lane >> 4, col = cb + n;
  float bv = bias1[col] + bias2[col];
#pragma unroll
  for (int v = 0; v < 8; ++v)
    OutF[(size_t)(row0 + hi*8 + v) * NO + col] = acc[v] + bv;
}

// ---------------------------------------------------------------------------
// skip += dot(xx_row, skip1_w) + skip1_b (broadcast over T). 1 block per row.
// ---------------------------------------------------------------------------
__global__ void tdg_skip1(const float* __restrict__ XX, const float* __restrict__ w,
                          const float* __restrict__ b, float* __restrict__ SKIP) {
  int row = blockIdx.x, tid = threadIdx.x;     // blockDim = 128
  __shared__ float red[128];
  float v = 0.f;
  if (tid < Tn) v = XX[(size_t)row * Tn + tid] * w[tid];
  red[tid] = v;
  __syncthreads();
  for (int s = 64; s > 0; s >>= 1) {
    if (tid < s) red[tid] += red[tid + s];
    __syncthreads();
  }
  float sc = red[0] + b[0];
  if (tid < Tn) SKIP[(size_t)row * Tn + tid] += sc;
}

// ---------------------------------------------------------------------------
// h = xg@msgT + msg_b ; gate = sigmoid(xg@gateT + gate_b). wave0->H, wave1->GATE.
// ---------------------------------------------------------------------------
__global__ void tdg_hgate(const _Float16* __restrict__ XG,
                          const _Float16* __restrict__ msgT, const _Float16* __restrict__ gateT,
                          const float* __restrict__ msg_b, const float* __restrict__ gate_b,
                          float* __restrict__ H, float* __restrict__ GATE) {
  constexpr int KD = 64;
  __shared__ alignas(32) _Float16 sA[16 * KD];
  __shared__ alignas(32) _Float16 sB[2 * KD * 16];
  int row0 = blockIdx.x * 16;
  int tid = threadIdx.x, lane = tid & 31, wave = tid >> 5;   // blockDim = 64
  for (int i = tid; i < 16 * KD; i += 64) {
    int r = i / KD, k = i - r * KD;
    sA[tdg_aswz(r, k)] = XG[(size_t)(row0 + r) * KD + k];
  }
  for (int i = tid; i < KD * 16; i += 64) { sB[i] = msgT[i]; sB[KD*16 + i] = gateT[i]; }
  __syncthreads();
  v8f acc = {};
  const _Float16* myB = sB + wave * KD * 16;
#pragma unroll
  for (int chunk = 0; chunk < KD / 32; ++chunk) {
    v16h a = tdg_frag_a(sA, chunk, lane);
    v16h b = tdg_frag_b(myB, 16, chunk, 0, lane);
    acc = __builtin_amdgcn_wmma_f32_16x16x32_f16(false, a, false, b, (short)0, acc, false, false);
  }
  int n = lane & 15, hi = lane >> 4;
  float bv = wave ? gate_b[n] : msg_b[n];
  float* dst = wave ? GATE : H;
#pragma unroll
  for (int v = 0; v < 8; ++v) {
    float val = acc[v] + bv;
    if (wave) val = 1.f / (1.f + expf(-val));
    dst[(size_t)(row0 + hi*8 + v) * DG + n] = val;
  }
}

__global__ void tdg_zero(float* __restrict__ p, size_t n) {
  size_t i = (size_t)blockIdx.x * blockDim.x + threadIdx.x;
  if (i < n) p[i] = 0.f;
}

// Backward aggregation: scatter h[b,n]*mask into AGG[b,cols[n,k]]. 1 block per (b,n).
__global__ void tdg_scatter(const float* __restrict__ H, const int* __restrict__ cols,
                            const float* __restrict__ maskv, float* __restrict__ AGG) {
  int row = blockIdx.x;                       // b*N + n
  int b = row >> 11, n = row & (Nn - 1);
  int tid = threadIdx.x;                      // blockDim = 320 = K*DG
  __shared__ float hr[DG];
  if (tid < DG) hr[tid] = H[(size_t)row * DG + tid];
  __syncthreads();
  int k = tid >> 4, d = tid & 15;
  float m = maskv[n * Kk + k];
  if (m != 0.f) {
    int c = cols[n * Kk + k];
    atomicAdd(&AGG[((size_t)b * Nn + c) * DG + d], m * hr[d]);
  }
}

// Forward gather + combine: gout = gate * (agg_f + agg_b), stored as f16.
__global__ void tdg_gout(const float* __restrict__ H, const float* __restrict__ GATE,
                         const float* __restrict__ AGG, const int* __restrict__ cols,
                         const float* __restrict__ maskv, _Float16* __restrict__ GOUT) {
  size_t i = (size_t)blockIdx.x * blockDim.x + threadIdx.x;
  if (i >= ROWSC * DG) return;
  int d = (int)(i & 15);
  size_t row = i >> 4;
  int b = (int)(row >> 11), n = (int)(row & (Nn - 1));
  const int*   cr = cols + n * Kk;
  const float* mr = maskv + n * Kk;
  size_t bbase = (size_t)b * Nn * DG;
  float aggf = 0.f;
#pragma unroll
  for (int k = 0; k < Kk; ++k) aggf += mr[k] * H[bbase + (size_t)cr[k] * DG + d];
  GOUT[i] = (_Float16)(GATE[i] * (aggf + AGG[i]));
}

// Batch stats: mean and rstd over [N,T] per batch.
__global__ void tdg_stats(const float* __restrict__ XX, float* __restrict__ stats) {
  int b = blockIdx.x, tid = threadIdx.x;      // blockDim = 256
  const size_t n = (size_t)Nn * Tn;
  const float* p = XX + (size_t)b * n;
  float s = 0.f, sq = 0.f;
  for (size_t i = tid; i < n; i += 256) { float v = p[i]; s += v; sq += v * v; }
  __shared__ float rs[256], rq[256];
  rs[tid] = s; rq[tid] = sq;
  __syncthreads();
  for (int st = 128; st > 0; st >>= 1) {
    if (tid < st) { rs[tid] += rs[tid + st]; rq[tid] += rq[tid + st]; }
    __syncthreads();
  }
  if (tid == 0) {
    float mean = rs[0] / (float)n;
    float var  = rq[0] / (float)n - mean * mean;
    stats[b] = mean;
    stats[32 + b] = rsqrtf(var + 1e-5f);
  }
}

__global__ void tdg_norm(float* __restrict__ XX, const float* __restrict__ stats,
                         const float* __restrict__ lnw, const float* __restrict__ lnb) {
  size_t i = (size_t)blockIdx.x * blockDim.x + threadIdx.x;
  if (i >= ROWSC * Tn) return;
  int b = (int)(i / ((size_t)Nn * Tn));
  size_t nt = i - (size_t)b * Nn * Tn;
  XX[i] = (XX[i] - stats[b]) * stats[32 + b] * lnw[nt] + lnb[nt];
}

// Final head: out = relu(relu(skip) @ lin3T + b3) @ w4 + b4  (one 16-row block).
__global__ void tdg_final(const float* __restrict__ SKIP, const _Float16* __restrict__ lin3T,
                          const float* __restrict__ b3, const float* __restrict__ w4,
                          const float* __restrict__ b4, float* __restrict__ out) {
  constexpr int KD = 96, NO = 64, NW = 4;
  __shared__ alignas(32) _Float16 sA[16 * KD];
  __shared__ alignas(32) _Float16 sB[KD * NO];
  __shared__ float sC[16 * NO];
  int row0 = blockIdx.x * 16;
  int tid = threadIdx.x, lane = tid & 31, wave = tid >> 5;   // blockDim = 128
  for (int i = tid; i < 16 * KD; i += NW * 32) {
    int r = i / KD, k = i - r * KD;
    sA[tdg_aswz(r, k)] = (_Float16)fmaxf(SKIP[(size_t)(row0 + r) * KD + k], 0.f);
  }
  for (int i = tid; i < KD * NO; i += NW * 32) sB[i] = lin3T[i];
  __syncthreads();
  v8f acc = {};
  int cb = wave * 16;
#pragma unroll
  for (int chunk = 0; chunk < KD / 32; ++chunk) {
    v16h a = tdg_frag_a(sA, chunk, lane);
    v16h b = tdg_frag_b(sB, NO, chunk, cb, lane);
    acc = __builtin_amdgcn_wmma_f32_16x16x32_f16(false, a, false, b, (short)0, acc, false, false);
  }
  int n = lane & 15, hi = lane >> 4, col = cb + n;
  float bv = b3[col];
#pragma unroll
  for (int v = 0; v < 8; ++v) sC[(hi*8 + v) * NO + col] = fmaxf(acc[v] + bv, 0.f);
  __syncthreads();
  if (tid < 16) {
    float y = b4[0];
#pragma unroll 8
    for (int o = 0; o < NO; ++o) y += sC[tid * NO + o] * w4[o];
    out[row0 + tid] = y;
  }
}

// ---------------------------------------------------------------------------
// Launch sequence
// ---------------------------------------------------------------------------
extern "C" void kernel_launch(void* const* d_in, const int* in_sizes, int n_in,
                              void* d_out, int out_size, void* d_ws, size_t ws_size,
                              hipStream_t stream) {
  (void)in_sizes; (void)n_in; (void)out_size; (void)ws_size;
  const float* x       = (const float*)d_in[0];
  const float* emb1    = (const float*)d_in[1];
  const float* emb2    = (const float*)d_in[2];
  const float* gc_w1   = (const float*)d_in[3];
  const float* gc_b1   = (const float*)d_in[4];
  const float* gc_w2   = (const float*)d_in[5];
  const float* gc_b2   = (const float*)d_in[6];
  const float* skip0_w = (const float*)d_in[7];
  const float* skip0_b = (const float*)d_in[8];
  const float* skip1_w = (const float*)d_in[9];
  const float* skip1_b = (const float*)d_in[10];
  const float* e1_w    = (const float*)d_in[11];
  const float* e1_b    = (const float*)d_in[12];
  const float* seas_w  = (const float*)d_in[13];
  const float* seas_b  = (const float*)d_in[14];
  const float* trend_w = (const float*)d_in[15];
  const float* trend_b = (const float*)d_in[16];
  const float* lin1_w  = (const float*)d_in[17];
  const float* lin1_b  = (const float*)d_in[18];
  const float* lin2_w  = (const float*)d_in[19];
  const float* lin2_b  = (const float*)d_in[20];
  const float* lin3_w  = (const float*)d_in[21];
  const float* lin3_b  = (const float*)d_in[22];
  const float* lin4_w  = (const float*)d_in[23];
  const float* lin4_b  = (const float*)d_in[24];
  const float* msg_w   = (const float*)d_in[25];
  const float* msg_b   = (const float*)d_in[26];
  const float* gate_w  = (const float*)d_in[27];
  const float* gate_b  = (const float*)d_in[28];
  const float* ln_w    = (const float*)d_in[29];
  const float* ln_b    = (const float*)d_in[30];

  char* ws = (char*)d_ws;
  _Float16* SEAS16 = (_Float16*)(ws + OFF_SEAS16);
  _Float16* TRS16  = (_Float16*)(ws + OFF_TRS16);
  float*    XX     = (float*)(ws + OFF_XX);
  float*    SKIP   = (float*)(ws + OFF_SKIP);
  float*    NV1    = (float*)(ws + OFF_NV1);
  float*    NV2    = (float*)(ws + OFF_NV2);
  int*      COLS   = (int*)(ws + OFF_COLS);
  float*    MASKV  = (float*)(ws + OFF_MASK);
  _Float16* WP     = (_Float16*)(ws + OFF_W);
  float*    STATS  = (float*)(ws + OFF_STATS);
  _Float16* XG16   = (_Float16*)(ws + OFF_XG16);
  _Float16* GOUT16 = (_Float16*)(ws + OFF_GOUT);
  float*    Hb     = (float*)(ws + OFF_H);
  float*    GATEb  = (float*)(ws + OFF_GATE);
  float*    AGGb   = (float*)(ws + OFF_AGG);

  // 1) weight prep (f16, fragment-order Bt layouts)
  auto wlaunch = [&](const float* src, size_t off, int O, int srcK, int dstK) {
    int n = O * dstK;
    hipLaunchKernelGGL(tdg_wprep, dim3((n + 255) / 256), dim3(256), 0, stream,
                       src, WP + off, O, srcK, dstK);
  };
  wlaunch(skip0_w, W_SKIP0, 96, 96, 96);
  wlaunch(seas_w,  W_SEAS,  96, 96, 96);
  wlaunch(trend_w, W_TREND, 96, 96, 96);
  wlaunch(e1_w,    W_E1,    96, 96, 96);
  wlaunch(lin1_w,  W_LIN1,  64, 96, 96);
  wlaunch(lin3_w,  W_LIN3,  64, 96, 96);
  wlaunch(lin2_w,  W_LIN2,  96, 16, 32);   // zero-padded K 16->32
  wlaunch(msg_w,   W_MSG,   16, 64, 64);
  wlaunch(gate_w,  W_GATE,  16, 64, 64);

  // 2) graph construction
  hipLaunchKernelGGL(tdg_nv, dim3((Nn*DN + 255)/256), dim3(256), 0, stream, emb1, gc_w1, gc_b1, NV1);
  hipLaunchKernelGGL(tdg_nv, dim3((Nn*DN + 255)/256), dim3(256), 0, stream, emb2, gc_w2, gc_b2, NV2);
  hipLaunchKernelGGL(tdg_topk, dim3(Nn), dim3(256), 0, stream, NV1, NV2, COLS, MASKV);

  // 3) series decomposition
  hipLaunchKernelGGL(tdg_decomp, dim3((unsigned)ROWSC), dim3(Tn), 0, stream, x, SEAS16, TRS16);

  // 4) skip = inp @ skip0^T + b
  hipLaunchKernelGGL((tdg_gemm16<96,96,96,0,0>), dim3((unsigned)(ROWSC/16)), dim3(192), 0, stream,
                     (const void*)x, WP + W_SKIP0, skip0_b, SKIP, (_Float16*)nullptr, (const float*)nullptr);

  // 5) xx = seas@Ws + trs@Wt + biases
  hipLaunchKernelGGL(tdg_xx_gemm, dim3((unsigned)(ROWSC/16)), dim3(192), 0, stream,
                     SEAS16, WP + W_SEAS, TRS16, WP + W_TREND, seas_b, trend_b, XX);

  // 6) skip += xx @ skip1_w + skip1_b (broadcast)
  hipLaunchKernelGGL(tdg_skip1, dim3((unsigned)ROWSC), dim3(128), 0, stream, XX, skip1_w, skip1_b, SKIP);

  // 7) xg = xx @ lin1^T + b  -> f16 (aliases SEAS16, now dead)
  hipLaunchKernelGGL((tdg_gemm16<96,96,64,0,2>), dim3((unsigned)(ROWSC/16)), dim3(128), 0, stream,
                     (const void*)XX, WP + W_LIN1, lin1_b, (float*)nullptr, XG16, (const float*)nullptr);

  // 8) h / gate (aliases TRS16, now dead)
  hipLaunchKernelGGL(tdg_hgate, dim3((unsigned)(ROWSC/16)), dim3(64), 0, stream,
                     XG16, WP + W_MSG, WP + W_GATE, msg_b, gate_b, Hb, GATEb);

  // 9) gated aggregation over both edge directions
  hipLaunchKernelGGL(tdg_zero, dim3((unsigned)((ROWSC*DG + 255)/256)), dim3(256), 0, stream,
                     AGGb, ROWSC * DG);
  hipLaunchKernelGGL(tdg_scatter, dim3((unsigned)ROWSC), dim3(Kk*DG), 0, stream, Hb, COLS, MASKV, AGGb);
  hipLaunchKernelGGL(tdg_gout, dim3((unsigned)((ROWSC*DG + 255)/256)), dim3(256), 0, stream,
                     Hb, GATEb, AGGb, COLS, MASKV, GOUT16);

  // 10) xx = gout @ lin2^T + b + residual (in-place over XX, K padded 16->32)
  hipLaunchKernelGGL((tdg_gemm16<32,16,96,1,3>), dim3((unsigned)(ROWSC/16)), dim3(192), 0, stream,
                     (const void*)GOUT16, WP + W_LIN2, lin2_b, XX, (_Float16*)nullptr, x);

  // 11) batch layernorm with elementwise [N,T] affine
  hipLaunchKernelGGL(tdg_stats, dim3(Bn), dim3(256), 0, stream, XX, STATS);
  hipLaunchKernelGGL(tdg_norm, dim3((unsigned)((ROWSC*Tn + 255)/256)), dim3(256), 0, stream,
                     XX, STATS, ln_w, ln_b);

  // 12) skip += xx @ e1^T + b
  hipLaunchKernelGGL((tdg_gemm16<96,96,96,0,1>), dim3((unsigned)(ROWSC/16)), dim3(192), 0, stream,
                     (const void*)XX, WP + W_E1, e1_b, SKIP, (_Float16*)nullptr, (const float*)nullptr);

  // 13) head: relu -> lin3 -> relu -> lin4
  hipLaunchKernelGGL(tdg_final, dim3((unsigned)(ROWSC/16)), dim3(128), 0, stream,
                     SKIP, WP + W_LIN3, lin3_b, lin4_w, lin4_b, (float*)d_out);
}
